// SCA_15358803050584
// MI455X (gfx1250) — matrix operand order repaired and checked by reading, not compiled, not run on hardware
//
#include <hip/hip_runtime.h>
#include <math.h>

typedef __attribute__((ext_vector_type(16))) _Float16 v16h;
typedef __attribute__((ext_vector_type(8)))  _Float16 v8h;
typedef __attribute__((ext_vector_type(4)))  _Float16 v4h;
typedef __attribute__((ext_vector_type(8)))  float    v8f;

#define BATCH 4

// ---- WMMA f16 16x16x32 fragment maps (CDNA5 ISA 7.12.2, wave32) ----
// A (16x32, M=lane&15): halves 0-7 -> K = 8*up + 0..7 ; halves 8-15 -> K = 16 + 8*up + 0..7
// B (32x16, N=lane&15): halves 0-7 -> K = 8*up + 0..7 ; halves 8-15 -> K = 16 + 8*up + 0..7
// C/D (16x16 f32): reg r -> M = r + 8*up ; N = lane&15
// Both runs are 16B-contiguous -> two ds_load_b128 per fragment (LDS rows 16B aligned).
__device__ __forceinline__ v16h frag_a(const _Float16* row, int up) {
  v8h lo = *(const v8h*)(row + 8 * up);
  v8h hi = *(const v8h*)(row + 16 + 8 * up);
  return __builtin_shufflevector(lo, hi, 0,1,2,3,4,5,6,7,8,9,10,11,12,13,14,15);
}
__device__ __forceinline__ v16h frag_b(const _Float16* row, int up) {
  v8h lo = *(const v8h*)(row + 8 * up);
  v8h hi = *(const v8h*)(row + 16 + 8 * up);
  return __builtin_shufflevector(lo, hi, 0,1,2,3,4,5,6,7,8,9,10,11,12,13,14,15);
}
__device__ __forceinline__ v8f wmma_f16(v16h a, v16h b, v8f c) {
  return __builtin_amdgcn_wmma_f32_16x16x32_f16(false, a, false, b, (short)0, c, false, false);
}
// XOR butterfly within 16-lane halves via ds_swizzle (group-of-32: and=0x1f, xor=mask<<10)
template <int IMM>
__device__ __forceinline__ float swz(float v) {
  return __int_as_float(__builtin_amdgcn_ds_swizzle(__float_as_int(v), IMM));
}
__device__ __forceinline__ float rowmax16(float v) {
  v = fmaxf(v, swz<0x041F>(v)); v = fmaxf(v, swz<0x081F>(v));
  v = fmaxf(v, swz<0x101F>(v)); v = fmaxf(v, swz<0x201F>(v));
  return v;
}
__device__ __forceinline__ float rowsum16(float v) {
  v += swz<0x041F>(v); v += swz<0x081F>(v);
  v += swz<0x101F>(v); v += swz<0x201F>(v);
  return v;
}

// ============================ Stage kernels ============================

// Channel mean & max per pixel -> smap[b][0][n]=avg, smap[b][1][n]=max
template <int C, int N>
__global__ void k_sa_maps(const float* __restrict__ x, float* __restrict__ smap) {
  int idx = blockIdx.x * blockDim.x + threadIdx.x;
  if (idx >= BATCH * N / 4) return;
  int b = idx / (N / 4), n4 = (idx - b * (N / 4)) * 4;
  const float* xp = x + (size_t)b * C * N + n4;
  float s0=0,s1=0,s2=0,s3=0, m0=-INFINITY,m1=-INFINITY,m2=-INFINITY,m3=-INFINITY;
#pragma unroll 4
  for (int c = 0; c < C; ++c) {
    const float4 v = *(const float4*)(xp + (size_t)c * N);
    s0+=v.x; s1+=v.y; s2+=v.z; s3+=v.w;
    m0=fmaxf(m0,v.x); m1=fmaxf(m1,v.y); m2=fmaxf(m2,v.z); m3=fmaxf(m3,v.w);
  }
  const float inv = 1.0f / (float)C;
  *(float4*)(smap + (size_t)b * 2 * N + n4)     = make_float4(s0*inv, s1*inv, s2*inv, s3*inv);
  *(float4*)(smap + (size_t)b * 2 * N + N + n4) = make_float4(m0, m1, m2, m3);
}

// 7x7 conv (2->1, pad 3) on smap, sigmoid, gate: x1 = sigmoid(conv)*x
template <int C, int H, int W>
__global__ void k_sa_conv(const float* __restrict__ x, const float* __restrict__ smap,
                          const float* __restrict__ wsa, float* __restrict__ x1) {
  constexpr int N = H * W;
  int idx = blockIdx.x * blockDim.x + threadIdx.x;
  if (idx >= BATCH * N) return;
  int b = idx / N, n = idx - b * N;
  int y = n / W, xx = n - y * W;
  float acc = 0.f;
#pragma unroll
  for (int i = 0; i < 2; ++i) {
    for (int ky = 0; ky < 7; ++ky) {
      int yy = y + ky - 3; if (yy < 0 || yy >= H) continue;
      for (int kx = 0; kx < 7; ++kx) {
        int xc = xx + kx - 3; if (xc < 0 || xc >= W) continue;
        acc += smap[((size_t)b * 2 + i) * N + yy * W + xc] * wsa[i * 49 + ky * 7 + kx];
      }
    }
  }
  float g = 1.f / (1.f + __expf(-acc));
  const float* xp = x  + (size_t)b * C * N + n;
  float*       op = x1 + (size_t)b * C * N + n;
#pragma unroll 4
  for (int c = 0; c < C; ++c) op[(size_t)c * N] = g * xp[(size_t)c * N];
}

// QKV projection as a WMMA GEMM: rows o<D -> Q, o<2D -> K, else -> V. J=2D+C is a
// multiple of 16 for all stages. Outputs are f16: Q,K [B][N][D]; V [B][C][N].
template <int C, int D, int N>
__global__ void k_qkv_gemm(const float* __restrict__ x1,
                           const float* __restrict__ wq, const float* __restrict__ bq,
                           const float* __restrict__ wk, const float* __restrict__ bk,
                           const float* __restrict__ wv, const float* __restrict__ bv,
                           _Float16* __restrict__ Q, _Float16* __restrict__ K,
                           _Float16* __restrict__ V) {
  alignas(16) __shared__ _Float16 sA[16][32];
  alignas(16) __shared__ _Float16 sB[4][16][32];   // [wave][n][k]
  int tid = threadIdx.x, wid = tid >> 5, lane = tid & 31;
  int lane15 = lane & 15, up = (lane >> 4) & 1;
  int b = blockIdx.z, o0 = blockIdx.y * 16, nblk = blockIdx.x * 64;
  int n0 = nblk + wid * 16;
  v8f acc = {};
  for (int k0 = 0; k0 < C; k0 += 32) {
    __syncthreads();
    { // stage A (weights, row-dispatched): 128 quads, 1 per thread
      int e = tid;
      int m = e >> 3, k4 = (e & 7) * 4;
      int o = o0 + m;
      const float* wrow = (o < D)     ? (wq + (size_t)o * C)
                        : (o < 2 * D) ? (wk + (size_t)(o - D) * C)
                                      : (wv + (size_t)(o - 2 * D) * C);
      const float4 w4 = *(const float4*)(wrow + k0 + k4);
      v4h h4 = {(_Float16)w4.x, (_Float16)w4.y, (_Float16)w4.z, (_Float16)w4.w};
      *(v4h*)&sA[m][k4] = h4;
    }
    for (int e = tid; e < 4 * 32 * 16 / 4; e += 128) { // stage B (x1 tile, transposed)
      int n4 = (e & 3) * 4, k = (e >> 2) & 31, w = e >> 7;
      const float4 v4 = *(const float4*)&x1[((size_t)b * C + k0 + k) * N + nblk + w * 16 + n4];
      sB[w][n4 + 0][k] = (_Float16)v4.x;
      sB[w][n4 + 1][k] = (_Float16)v4.y;
      sB[w][n4 + 2][k] = (_Float16)v4.z;
      sB[w][n4 + 3][k] = (_Float16)v4.w;
    }
    __syncthreads();
    acc = wmma_f16(frag_a(&sA[lane15][0], up), frag_b(&sB[wid][lane15][0], up), acc);
  }
#pragma unroll
  for (int r = 0; r < 8; ++r) {
    int o = o0 + r + 8 * up, n = n0 + lane15;
    float val = acc[r];
    if (o < D)          Q[((size_t)b * N + n) * D + o]           = (_Float16)(val + bq[o]);
    else if (o < 2 * D) K[((size_t)b * N + n) * D + (o - D)]     = (_Float16)(val + bk[o - D]);
    else                V[((size_t)b * C + (o - 2 * D)) * N + n] = (_Float16)(val + bv[o - 2 * D]);
  }
}

// Flash attention + gate: x2[b][c][n] = (softmax(Q K^T) @ V^T)[n][c] * x1[b][c][n].
// Block: 256 threads (8 waves) = 2 query tiles x 4 channel-waves sharing staged K/V.
// Q/K/V are f16 in global -> staging is pure wide copies, no converts in the loop.
template <int C, int D, int N>
__global__ void k_flash(const _Float16* __restrict__ Q, const _Float16* __restrict__ K,
                        const _Float16* __restrict__ V, const float* __restrict__ x1,
                        _Float16* __restrict__ x2) {
  constexpr int KS = (D + 31) / 32;
  constexpr int DP = KS * 32;       // padded K-dim for QK^T
  constexpr int DH8 = D / 8;        // 16-byte chunks per row
  constexpr int CS = C / 4;         // channel slice per wave
  constexpr int NA = CS / 16;       // accumulators per wave

  alignas(16) __shared__ _Float16 sQ[2][16][DP];
  alignas(16) __shared__ _Float16 sK[16][DP];
  alignas(16) __shared__ _Float16 sVt[C][32];   // [channel][key(16) + 16 zero pad]
  alignas(16) __shared__ _Float16 sP[8][16][32];

  int tid = threadIdx.x, wid = tid >> 5, lane = tid & 31;
  int lane15 = lane & 15, up = (lane >> 4) & 1;
  int qt = wid >> 2, cw = wid & 3;
  int b = blockIdx.y, nb = blockIdx.x * 32, n0 = nb + qt * 16;

  // Zero the pad regions once (they are never rewritten).
  if constexpr (DP > D) {
    for (int e = tid; e < 16 * (DP - D); e += 256) {
      int m = e / (DP - D), i = D + e % (DP - D);
      sQ[0][m][i] = (_Float16)0.f; sQ[1][m][i] = (_Float16)0.f; sK[m][i] = (_Float16)0.f;
    }
  }
  for (int e = tid; e < 8 * 16 * 16; e += 256) {
    sP[e >> 8][(e >> 4) & 15][16 + (e & 15)] = (_Float16)0.f;
  }
  for (int e = tid; e < C * 16; e += 256) sVt[e >> 4][16 + (e & 15)] = (_Float16)0.f;

  // Stage both Q tiles [16][D] (straight f16 copies, 16B chunks)
  for (int e = tid; e < 2 * 16 * DH8; e += 256) {
    int q2 = e / (16 * DH8), rem = e % (16 * DH8);
    int m = rem / DH8, i8 = (rem % DH8) * 8;
    *(v8h*)&sQ[q2][m][i8] = *(const v8h*)&Q[((size_t)b * N + nb + q2 * 16 + m) * D + i8];
  }
  __syncthreads();

  v16h aQ[KS];
#pragma unroll
  for (int s = 0; s < KS; ++s) aQ[s] = frag_a(&sQ[qt][lane15][32 * s], up);

  v8f acc[NA];
#pragma unroll
  for (int a = 0; a < NA; ++a) acc[a] = {};
  float m_r[8], l_r[8];
#pragma unroll
  for (int r = 0; r < 8; ++r) { m_r[r] = -INFINITY; l_r[r] = 0.f; }

  constexpr int NT = N / 16;
  for (int j = 0; j < NT; ++j) {
    int nj = j * 16;
    __syncthreads();  // previous iteration done with sK/sVt
    for (int e = tid; e < 16 * DH8; e += 256) {  // K tile [key][D]
      int k = e / DH8, i8 = (e % DH8) * 8;
      *(v8h*)&sK[k][i8] = *(const v8h*)&K[((size_t)b * N + nj + k) * D + i8];
    }
    for (int e = tid; e < 2 * C; e += 256) {  // V tile transposed [channel][key]
      int c = e >> 1, k8 = (e & 1) * 8;
      *(v8h*)&sVt[c][k8] = *(const v8h*)&V[((size_t)b * C + c) * N + nj + k8];
    }
    { // prefetch next V tile into cache (global_prefetch_b8)
      int jn = (j + 1 < NT) ? nj + 16 : nj;
      for (int c = tid; c < C; c += 256)
        __builtin_prefetch(&V[((size_t)b * C + c) * N + jn], 0, 0);
    }
    __syncthreads();

    // S = Q[16,D] x K^T[D,16]  (K-dim zero-padded to DP)
    v8f s = {};
#pragma unroll
    for (int ks = 0; ks < KS; ++ks)
      s = wmma_f16(aQ[ks], frag_b(&sK[lane15][32 * ks], up), s);

    // Online softmax; rows live in 16-lane halves per the C/D layout.
    float alpha[8];
#pragma unroll
    for (int r = 0; r < 8; ++r) {
      float val = s[r];
      float mn = fmaxf(m_r[r], rowmax16(val));
      float p = __expf(val - mn);
      alpha[r] = __expf(m_r[r] - mn);
      l_r[r] = l_r[r] * alpha[r] + rowsum16(p);
      m_r[r] = mn;
      sP[wid][r + 8 * up][lane15] = (_Float16)p;   // re-layout P for the A fragment
    }
#pragma unroll
    for (int a = 0; a < NA; ++a)
#pragma unroll
      for (int r = 0; r < 8; ++r) acc[a][r] *= alpha[r];

    // sP is wave-private: wave-level LDS ordering is enough (split dep counters).
    asm volatile("s_wait_dscnt 0x0" ::: "memory");

    v16h aP = frag_a(&sP[wid][lane15][0], up);     // halves >=16 hit the zero pad
#pragma unroll
    for (int a = 0; a < NA; ++a) {
      int c = cw * CS + a * 16 + lane15;
      acc[a] = wmma_f16(aP, frag_b(&sVt[c][0], up), acc[a]);
    }
  }

  // Finalize: x2 = (O/l) * x1  (stored f16 for the translayer GEMM)
#pragma unroll
  for (int r = 0; r < 8; ++r) {
    float inv = 1.0f / l_r[r];
    int nn = n0 + r + 8 * up;
#pragma unroll
    for (int a = 0; a < NA; ++a) {
      int c = cw * CS + a * 16 + lane15;
      size_t off = ((size_t)b * C + c) * N + nn;
      x2[off] = (_Float16)((acc[a][r] * inv) * x1[off]);
    }
  }
}

// Translayer: out = relu(BN(w_t @ x2)), WMMA GEMM with fused BN+ReLU epilogue.
// x2 already f16 -> B staging is a pure transpose, no converts.
template <int C, int N>
__global__ void k_trans(const _Float16* __restrict__ x2, const float* __restrict__ wt,
                        const float* __restrict__ gamma, const float* __restrict__ beta,
                        const float* __restrict__ mean, const float* __restrict__ var,
                        float* __restrict__ out) {
  alignas(16) __shared__ _Float16 sA[16][32];
  alignas(16) __shared__ _Float16 sB[4][16][32];
  int tid = threadIdx.x, wid = tid >> 5, lane = tid & 31;
  int lane15 = lane & 15, up = (lane >> 4) & 1;
  int b = blockIdx.z, o0 = blockIdx.y * 16, nblk = blockIdx.x * 64;
  int n0 = nblk + wid * 16;
  v8f acc = {};
  for (int k0 = 0; k0 < C; k0 += 32) {
    __syncthreads();
    {
      int e = tid;
      int m = e >> 3, k4 = (e & 7) * 4;
      const float4 w4 = *(const float4*)&wt[(size_t)(o0 + m) * C + k0 + k4];
      v4h h4 = {(_Float16)w4.x, (_Float16)w4.y, (_Float16)w4.z, (_Float16)w4.w};
      *(v4h*)&sA[m][k4] = h4;
    }
    for (int e = tid; e < 4 * 32 * 16 / 4; e += 128) {
      int n4 = (e & 3) * 4, k = (e >> 2) & 31, w = e >> 7;
      const v4h v4 = *(const v4h*)&x2[((size_t)b * C + k0 + k) * N + nblk + w * 16 + n4];
      sB[w][n4 + 0][k] = v4[0];
      sB[w][n4 + 1][k] = v4[1];
      sB[w][n4 + 2][k] = v4[2];
      sB[w][n4 + 3][k] = v4[3];
    }
    __syncthreads();
    acc = wmma_f16(frag_a(&sA[lane15][0], up), frag_b(&sB[wid][lane15][0], up), acc);
  }
#pragma unroll
  for (int r = 0; r < 8; ++r) {
    int o = o0 + r + 8 * up;
    float sc = gamma[o] * rsqrtf(var[o] + 1e-5f);
    float sh = beta[o] - mean[o] * sc;
    float y = acc[r] * sc + sh;
    out[((size_t)b * C + o) * N + n0 + lane15] = fmaxf(y, 0.f);
  }
}

// ============================ Host side ============================

struct StageParams {
  const float *w_sa, *wq, *bq, *wk, *bk, *wv, *bv, *w_t, *gamma, *beta, *mean, *var;
};

static StageParams get_params(void* const* d_in, int base, bool sorted_inner) {
  auto f = [&](int k) { return (const float*)d_in[base + k]; };
  StageParams p;
  if (sorted_inner) {
    // jax tree-flatten (sorted keys): bk, bn_beta, bn_gamma, bn_mean, bn_var,
    //                                 bq, bv, w_sa, w_t, wk, wq, wv
    p.bk = f(0);  p.beta = f(1); p.gamma = f(2); p.mean = f(3); p.var = f(4);
    p.bq = f(5);  p.bv = f(6);   p.w_sa = f(7);  p.w_t = f(8);
    p.wk = f(9);  p.wq = f(10);  p.wv = f(11);
  } else {
    // dict insertion order: w_sa, wq, bq, wk, bk, wv, bv, w_t, gamma, beta, mean, var
    p.w_sa = f(0); p.wq = f(1); p.bq = f(2); p.wk = f(3); p.bk = f(4);
    p.wv = f(5);   p.bv = f(6); p.w_t = f(7);
    p.gamma = f(8); p.beta = f(9); p.mean = f(10); p.var = f(11);
  }
  return p;
}

template <int C, int H, int W>
static void run_stage(const float* x, const StageParams& p, float* out,
                      float* x1, _Float16* Qb, _Float16* Kb, _Float16* Vb,
                      _Float16* x2, float* smap, hipStream_t stream) {
  constexpr int N = H * W;
  constexpr int D = C / 8;
  constexpr int J = 2 * D + C;
  k_sa_maps<C, N><<<(BATCH * N / 4 + 255) / 256, 256, 0, stream>>>(x, smap);
  k_sa_conv<C, H, W><<<(BATCH * N + 255) / 256, 256, 0, stream>>>(x, smap, p.w_sa, x1);
  k_qkv_gemm<C, D, N><<<dim3(N / 64, J / 16, BATCH), 128, 0, stream>>>(
      x1, p.wq, p.bq, p.wk, p.bk, p.wv, p.bv, Qb, Kb, Vb);
  k_flash<C, D, N><<<dim3(N / 32, BATCH), 256, 0, stream>>>(Qb, Kb, Vb, x1, x2);
  k_trans<C, N><<<dim3(N / 64, C / 16, BATCH), 128, 0, stream>>>(
      x2, p.w_t, p.gamma, p.beta, p.mean, p.var, out);
}

extern "C" void kernel_launch(void* const* d_in, const int* in_sizes, int n_in,
                              void* d_out, int out_size, void* d_ws, size_t ws_size,
                              hipStream_t stream) {
  (void)out_size; (void)ws_size;
  // Detect flattening order of the nested param dicts from in_sizes.
  bool grouped = (n_in >= 5) && (in_sizes[1] == 4 * 128 * 32 * 32);
  int faIdx[4], pBase[4];
  bool sorted_inner;
  if (grouped) {
    for (int i = 0; i < 4; ++i) { faIdx[i] = i; pBase[i] = 4 + 12 * i; }
    sorted_inner = (in_sizes[4] == 8);
  } else {
    for (int i = 0; i < 4; ++i) { faIdx[i] = 13 * i; pBase[i] = 13 * i + 1; }
    sorted_inner = (in_sizes[1] == 8);
  }

  // Workspace layout (byte offsets, 16B-aligned regions), sized by stage 1.
  char* wsb = (char*)d_ws;
  float*    x1  = (float*)wsb;                                   // 4 MB f32
  _Float16* Vb  = (_Float16*)(wsb + (4u << 20));                 // 2 MB f16
  _Float16* x2  = (_Float16*)(wsb + (6u << 20));                 // 2 MB f16
  _Float16* Qb  = (_Float16*)(wsb + (8u << 20));                 // 256 KB f16
  _Float16* Kb  = (_Float16*)(wsb + (8u << 20) + (256u << 10));  // 256 KB f16
  float*    sm  = (float*)(wsb + (8u << 20) + (512u << 10));     // 128 KB f32

  float* out = (float*)d_out;
  const float* fa1 = (const float*)d_in[faIdx[0]];
  const float* fa2 = (const float*)d_in[faIdx[1]];
  const float* fa3 = (const float*)d_in[faIdx[2]];
  const float* fa4 = (const float*)d_in[faIdx[3]];
  StageParams p1 = get_params(d_in, pBase[0], sorted_inner);
  StageParams p2 = get_params(d_in, pBase[1], sorted_inner);
  StageParams p3 = get_params(d_in, pBase[2], sorted_inner);
  StageParams p4 = get_params(d_in, pBase[3], sorted_inner);

  size_t o0 = 0;
  run_stage< 64, 64, 64>(fa1, p1, out + o0, x1, Qb, Kb, Vb, x2, sm, stream);
  o0 += (size_t)4 * 64 * 64 * 64;
  run_stage<128, 32, 32>(fa2, p2, out + o0, x1, Qb, Kb, Vb, x2, sm, stream);
  o0 += (size_t)4 * 128 * 32 * 32;
  run_stage<256, 16, 16>(fa3, p3, out + o0, x1, Qb, Kb, Vb, x2, sm, stream);
  o0 += (size_t)4 * 256 * 16 * 16;
  run_stage<512,  8,  8>(fa4, p4, out + o0, x1, Qb, Kb, Vb, x2, sm, stream);
}
